// HybridQuantumLinear_46480136077942
// MI455X (gfx1250) — matrix operand order repaired and checked by reading, not compiled, not run on hardware
//
#include <hip/hip_runtime.h>
#include <hip/hip_fp16.h>

typedef __attribute__((ext_vector_type(16))) _Float16 v16h;
typedef __attribute__((ext_vector_type(8)))  _Float16 v8h;
typedef __attribute__((ext_vector_type(8)))  float    v8f;

#define DIN 512
#define DOUT 512
#define DIM 256            // 2^8 states
#define NQ 8
#define NL 4
#define MTILE 32           // batch rows per block
#define PI_F 3.14159265358979323846f

// ---------------------------------------------------------------------------
// Statevector helpers: 256 amplitudes per wave32 -> 8 complex amps per lane.
// amp index i = r*32 + lane. Bit b of i: b<5 -> lane bit, b>=5 -> register bit.
// Qubit positions are template parameters so register indices constant-fold.
// ---------------------------------------------------------------------------
template <int P>
__device__ __forceinline__ void apply1q(float (&ar)[8], float (&ai)[8], int lane,
                                        float g00r, float g00i, float g01r, float g01i,
                                        float g10r, float g10i, float g11r, float g11i) {
    if constexpr (P >= 5) {
        constexpr int RB = 1 << (P - 5);
#pragma unroll
        for (int r = 0; r < 8; ++r) {
            if ((r & RB) == 0) {
                const int r1 = r | RB;
                float a0r = ar[r], a0i = ai[r], a1r = ar[r1], a1i = ai[r1];
                ar[r]  = g00r*a0r - g00i*a0i + g01r*a1r - g01i*a1i;
                ai[r]  = g00r*a0i + g00i*a0r + g01r*a1i + g01i*a1r;
                ar[r1] = g10r*a0r - g10i*a0i + g11r*a1r - g11i*a1i;
                ai[r1] = g10r*a0i + g10i*a0r + g11r*a1i + g11i*a1r;
            }
        }
    } else {
        constexpr int M = 1 << P;
        const bool hi = (lane >> P) & 1;
        const float ur = hi ? g11r : g00r, ui = hi ? g11i : g00i;  // coeff of own amp
        const float vr = hi ? g10r : g01r, vi = hi ? g10i : g01i;  // coeff of partner
#pragma unroll
        for (int r = 0; r < 8; ++r) {
            float br = __shfl_xor(ar[r], M, 32);
            float bi = __shfl_xor(ai[r], M, 32);
            float nr = ur*ar[r] - ui*ai[r] + vr*br - vi*bi;
            float ni = ur*ai[r] + ui*ar[r] + vr*bi + vi*br;
            ar[r] = nr; ai[r] = ni;
        }
    }
}

template <int P>
__device__ __forceinline__ void applyrz(float (&ar)[8], float (&ai)[8], int lane,
                                        float ch, float sh) {
#pragma unroll
    for (int r = 0; r < 8; ++r) {
        const int i = r * 32 + lane;
        const float ei = ((i >> P) & 1) ? sh : -sh;   // diag(e^{-ih}, e^{+ih})
        float nr = ar[r]*ch - ai[r]*ei;
        float ni = ar[r]*ei + ai[r]*ch;
        ar[r] = nr; ai[r] = ni;
    }
}

template <int Q>  // control Q, target Q+1; control bit = 7-Q, target bit = 6-Q
__device__ __forceinline__ void applycnot(float (&ar)[8], float (&ai)[8], int lane) {
    constexpr int PC = 7 - Q, PT = 6 - Q;
    if constexpr (PT >= 5) {          // both register bits
        constexpr int CB = 1 << (PC - 5), TB = 1 << (PT - 5);
#pragma unroll
        for (int r = 0; r < 8; ++r) {
            if ((r & CB) && !(r & TB)) {
                const int r1 = r | TB;
                float t;
                t = ar[r]; ar[r] = ar[r1]; ar[r1] = t;
                t = ai[r]; ai[r] = ai[r1]; ai[r1] = t;
            }
        }
    } else if constexpr (PC >= 5) {   // control = reg bit 0, target = lane bit 4
#pragma unroll
        for (int r = 0; r < 8; ++r) {
            if (r & 1) {
                ar[r] = __shfl_xor(ar[r], 16, 32);
                ai[r] = __shfl_xor(ai[r], 16, 32);
            }
        }
    } else {                          // both lane bits
        constexpr int CM = 1 << PC, TM = 1 << PT;
        const bool cs = lane & CM;
#pragma unroll
        for (int r = 0; r < 8; ++r) {
            float br = __shfl_xor(ar[r], TM, 32);
            float bi = __shfl_xor(ai[r], TM, 32);
            if (cs) { ar[r] = br; ai[r] = bi; }
        }
    }
}

template <int Q>
__device__ __forceinline__ void rot3(const float* th, float (&ar)[8], float (&ai)[8], int lane) {
    constexpr int P = 7 - Q;
    float h, c, s;
    h = 0.5f * th[0]; c = __cosf(h); s = __sinf(h);               // RX
    apply1q<P>(ar, ai, lane, c, 0.f, 0.f, -s, 0.f, -s, c, 0.f);
    h = 0.5f * th[1]; c = __cosf(h); s = __sinf(h);               // RY
    apply1q<P>(ar, ai, lane, c, 0.f, -s, 0.f, s, 0.f, c, 0.f);
    h = 0.5f * th[2]; c = __cosf(h); s = __sinf(h);               // RZ
    applyrz<P>(ar, ai, lane, c, s);
}

// ---------------------------------------------------------------------------
// Kernel 1: build the fixed 256x256 unitary U(theta), column j = circuit(|j>).
// Store as B-matrix for WMMA: Wstore[2n+c][k] = c?Im:Re of U[n][k] (f16, N x K).
// ---------------------------------------------------------------------------
__global__ __launch_bounds__(32) void build_u_kernel(const float* __restrict__ theta,
                                                     _Float16* __restrict__ wstore) {
    const int j = blockIdx.x;        // column of U
    const int lane = threadIdx.x;
    float ar[8], ai[8];
#pragma unroll
    for (int r = 0; r < 8; ++r) { ar[r] = (r * 32 + lane == j) ? 1.f : 0.f; ai[r] = 0.f; }

    for (int l = 0; l < NL; ++l) {
        const float* th = theta + l * (NQ * 3);
        rot3<0>(th + 0,  ar, ai, lane);
        rot3<1>(th + 3,  ar, ai, lane);
        rot3<2>(th + 6,  ar, ai, lane);
        rot3<3>(th + 9,  ar, ai, lane);
        rot3<4>(th + 12, ar, ai, lane);
        rot3<5>(th + 15, ar, ai, lane);
        rot3<6>(th + 18, ar, ai, lane);
        rot3<7>(th + 21, ar, ai, lane);
        if (l & 1) {
            applycnot<1>(ar, ai, lane); applycnot<3>(ar, ai, lane); applycnot<5>(ar, ai, lane);
        } else {
            applycnot<0>(ar, ai, lane); applycnot<2>(ar, ai, lane);
            applycnot<4>(ar, ai, lane); applycnot<6>(ar, ai, lane);
        }
    }
#pragma unroll
    for (int r = 0; r < 8; ++r) {
        const int i = r * 32 + lane;
        wstore[(size_t)(2 * i)     * DIM + j] = (_Float16)ar[r];
        wstore[(size_t)(2 * i + 1) * DIM + j] = (_Float16)ai[r];
    }
}

// ---------------------------------------------------------------------------
// Kernel 2: enc = tanh(x @ W_in^T) * pi, then psi0 row = product state (f16).
// One wave per batch row; 8 rows / 256-thread block. Lane owns 8 contiguous
// amplitudes -> single b128 store per row segment.
// ---------------------------------------------------------------------------
__global__ __launch_bounds__(256) void encode_kernel(const float* __restrict__ x,
                                                     const float* __restrict__ w_in,
                                                     _Float16* __restrict__ psi0) {
    __shared__ float wlds[NQ * DIN];
    const int tid = threadIdx.x;
    for (int i = tid; i < NQ * DIN; i += 256) wlds[i] = w_in[i];
    __syncthreads();

    const int wave = tid >> 5, lane = tid & 31;
    const int row = blockIdx.x * 8 + wave;
    const float* xr = x + (size_t)row * DIN;

    float acc[NQ];
#pragma unroll
    for (int q = 0; q < NQ; ++q) acc[q] = 0.f;
    for (int c = lane; c < DIN; c += 32) {
        const float xv = xr[c];
#pragma unroll
        for (int q = 0; q < NQ; ++q) acc[q] += xv * wlds[q * DIN + c];
    }
#pragma unroll
    for (int off = 16; off > 0; off >>= 1)
#pragma unroll
        for (int q = 0; q < NQ; ++q) acc[q] += __shfl_xor(acc[q], off, 32);

    float cq[NQ], sq[NQ];
#pragma unroll
    for (int q = 0; q < NQ; ++q) {
        const float h = 0.5f * PI_F * tanhf(acc[q]);   // enc/2
        cq[q] = __cosf(h); sq[q] = __sinf(h);
    }
    const int i0 = lane * 8;
    v8h vv;
#pragma unroll
    for (int e = 0; e < 8; ++e) {
        const int i = i0 + e;
        float v = 1.f;
#pragma unroll
        for (int q = 0; q < NQ; ++q) v *= ((i >> (7 - q)) & 1) ? sq[q] : cq[q];
        vv[e] = (_Float16)v;
    }
    *(v8h*)(psi0 + (size_t)row * DIM + i0) = vv;
}

// ---------------------------------------------------------------------------
// Kernel 3: WMMA GEMM  y = psi0 (Mx256 f16) @ W (256x512 f16, cols 2n/2n+1 =
// Re/Im of U row n), fused epilogue: probs=|y|^2, z = probs@Zsigns, out=z@W_out^T.
// Block = 256 threads (8 waves), M-tile = 32 rows, each wave owns 64 columns.
// Each B fragment feeds 2 WMMAs, each A fragment 4 -> latency-hiding ILP.
// ---------------------------------------------------------------------------
__global__ __launch_bounds__(256) void gemm_kernel(const _Float16* __restrict__ psi0,
                                                   const _Float16* __restrict__ wstore,
                                                   const float* __restrict__ w_out,
                                                   float* __restrict__ out) {
    __shared__ __align__(16) float probs[MTILE][DIM + 4];  // stride 260: rows 16B-aligned
    __shared__ float zl[MTILE][NQ];

    const int tid = threadIdx.x;
    const int wave = tid >> 5, lane = tid & 31;
    const int m = lane & 15, hi = lane >> 4;
    const int row0 = blockIdx.x * MTILE;

    v8f acc[2][4];
#pragma unroll
    for (int mi = 0; mi < 2; ++mi)
#pragma unroll
        for (int t = 0; t < 4; ++t) acc[mi][t] = (v8f)(0.f);

    // A fragment: 16x32 f16 (lane m = row; K = 16*hi + e, contiguous to match B)
    const _Float16* aBase0 = psi0 + (size_t)(row0 + m) * DIM + hi * 16;
    const _Float16* aBase1 = aBase0 + (size_t)16 * DIM;
    // B fragment 32x16: lane m = column n; K = 16*hi + e (one contiguous 32B run)
    const _Float16* bBase  = wstore + (size_t)(wave * 64 + m) * DIM + hi * 16;

#pragma unroll
    for (int kk = 0; kk < DIM; kk += 32) {
        const v8h a0l = *(const v8h*)(aBase0 + kk);
        const v8h a0h = *(const v8h*)(aBase0 + kk + 8);
        const v8h a1l = *(const v8h*)(aBase1 + kk);
        const v8h a1h = *(const v8h*)(aBase1 + kk + 8);
        const v16h A0 = __builtin_shufflevector(a0l, a0h, 0,1,2,3,4,5,6,7,8,9,10,11,12,13,14,15);
        const v16h A1 = __builtin_shufflevector(a1l, a1h, 0,1,2,3,4,5,6,7,8,9,10,11,12,13,14,15);
#pragma unroll
        for (int t = 0; t < 4; ++t) {
            const v16h B = *(const v16h*)(bBase + (size_t)t * 16 * DIM + kk);
            acc[0][t] = __builtin_amdgcn_wmma_f32_16x16x32_f16(
                false, A0, false, B, (short)0, acc[0][t], false, false);
            acc[1][t] = __builtin_amdgcn_wmma_f32_16x16x32_f16(
                false, A1, false, B, (short)0, acc[1][t], false, false);
        }
    }

    // |y|^2: Re in even columns (even lanes), Im in odd -> pair via shfl_xor(1)
#pragma unroll
    for (int mi = 0; mi < 2; ++mi)
#pragma unroll
        for (int t = 0; t < 4; ++t)
#pragma unroll
            for (int v = 0; v < 8; ++v) {
                const float own = acc[mi][t][v];
                const float par = __shfl_xor(own, 1, 32);
                const float pr = own * own + par * par;
                if ((lane & 1) == 0) {
                    const int i = wave * 32 + t * 8 + (m >> 1);   // amplitude 0..255
                    probs[mi * 16 + v + hi * 8][i] = pr;
                }
            }
    __syncthreads();

    // z[row][q] = sum_i probs[row][i] * (1 - 2*bit_{7-q}(i)); one (row,q)/thread
    {
        const int rrow = tid >> 3, q = tid & 7;
        const int shift = 7 - q;
        float z = 0.f;
        const float4* pr4 = (const float4*)probs[rrow];
#pragma unroll 8
        for (int i4 = 0; i4 < DIM / 4; ++i4) {
            const float4 p = pr4[i4];
            const int i = i4 * 4;
            z += (((i + 0) >> shift) & 1) ? -p.x : p.x;
            z += (((i + 1) >> shift) & 1) ? -p.y : p.y;
            z += (((i + 2) >> shift) & 1) ? -p.z : p.z;
            z += (((i + 3) >> shift) & 1) ? -p.w : p.w;
        }
        zl[rrow][q] = z;
    }
    __syncthreads();

    // out[row][d] = sum_q z[row][q] * W_out[d][q]   (32 x 512 per block)
#pragma unroll 4
    for (int it = 0; it < (MTILE * DOUT) / 256; ++it) {
        const int idx = it * 256 + tid;
        const int rrow = idx >> 9, d = idx & 511;
        const float4 w0 = *(const float4*)(w_out + (size_t)d * 8);
        const float4 w1 = *(const float4*)(w_out + (size_t)d * 8 + 4);
        const float* z = zl[rrow];
        float o = z[0]*w0.x + z[1]*w0.y + z[2]*w0.z + z[3]*w0.w
                + z[4]*w1.x + z[5]*w1.y + z[6]*w1.z + z[7]*w1.w;
        out[(size_t)(row0 + rrow) * DOUT + d] = o;
    }
}

// ---------------------------------------------------------------------------
extern "C" void kernel_launch(void* const* d_in, const int* in_sizes, int n_in,
                              void* d_out, int out_size, void* d_ws, size_t ws_size,
                              hipStream_t stream) {
    const float* x     = (const float*)d_in[0];   // (B, 512)
    const float* w_in  = (const float*)d_in[1];   // (8, 512)
    const float* w_out = (const float*)d_in[2];   // (512, 8)
    const float* theta = (const float*)d_in[3];   // (4, 8, 3)
    float* out = (float*)d_out;                   // (B, 512) f32

    const int B = in_sizes[0] / DIN;              // 8192

    _Float16* wstore = (_Float16*)d_ws;                               // 512*256 f16 = 256 KB
    _Float16* psi0   = (_Float16*)((char*)d_ws + (size_t)2 * DIM * DIM * sizeof(_Float16));

    build_u_kernel<<<DIM, 32, 0, stream>>>(theta, wstore);
    encode_kernel<<<B / 8, 256, 0, stream>>>(x, w_in, psi0);
    gemm_kernel<<<B / MTILE, 256, 0, stream>>>(psi0, wstore, w_out, out);
}
